// TransformerQuantizer_46703474376945
// MI455X (gfx1250) — matrix operand order repaired and checked by reading, not compiled, not run on hardware
//
#include <hip/hip_runtime.h>
#include <hip/hip_bf16.h>
#include <math.h>

typedef __attribute__((ext_vector_type(16))) _Float16 v16h;
typedef __attribute__((ext_vector_type(8)))  _Float16 h8;
typedef __attribute__((ext_vector_type(8)))  float    v8f;

#define C_DIM   512
#define S_DIM   1024
#define K_CODE  4096
#define DH      64
#define NS      16384          // N * S

#define ACT_NONE 0
#define ACT_GELU 1

// ---------------------------------------------------------------------------
__device__ __forceinline__ v16h frag_cat(h8 lo, h8 hi8) {
    return __builtin_shufflevector(lo, hi8, 0,1,2,3,4,5,6,7,8,9,10,11,12,13,14,15);
}

// 2D sinusoidal positional encoding, computed on the fly
__device__ __forceinline__ float pe_val(int s, int c) {
    int j, L;
    if (c < 256) { L = s >> 5;  j = c; }
    else         { L = s & 31;  j = c - 256; }
    int   p2   = (j >> 1) << 1;
    float freq = __expf(-(float)p2 * (9.210340371976184f / 256.0f)); // ln(1e4)
    float ang  = (float)L * freq;
    return (j & 1) ? __cosf(ang) : __sinf(ang);
}

// ---------------------------------------------------------------------------
// Weight preparation: f32 (K x N) -> f16 (N x K)  [WMMA B-fragment layout]
__global__ void cvt_transpose_w(const float* __restrict__ W, _Float16* __restrict__ o,
                                int Kd, int Ncol) {
    int idx = blockIdx.x * blockDim.x + threadIdx.x;   // coalesced along N
    int n = idx % Ncol, k = idx / Ncol;
    o[(size_t)n * Kd + k] = (_Float16)W[idx];
}
__global__ void cvt_plain(const float* __restrict__ W, _Float16* __restrict__ o) {
    int idx = blockIdx.x * blockDim.x + threadIdx.x;
    o[idx] = (_Float16)W[idx];
}

// ---------------------------------------------------------------------------
// tokens: x[n,s,c] = latent0[n,c,s] + PE[s,c]   (stored f16)
__global__ void embed_kernel(const float* __restrict__ lat, _Float16* __restrict__ x) {
    int idx = blockIdx.x * blockDim.x + threadIdx.x;   // over N*S*C
    int c   = idx & (C_DIM - 1);
    int row = idx >> 9;            // n*S + s
    int s   = row & (S_DIM - 1);
    int n   = row >> 10;
    x[idx] = (_Float16)(lat[((size_t)(n * C_DIM + c) << 10) + s] + pe_val(s, c));
}

// ---------------------------------------------------------------------------
// WMMA GEMM: C = act(alpha * A(MxK) @ Bt(NxK)^T + bias)
// A f16 row-major, Bt f16 (N x K) row-major -> pure h8 copies into LDS.
// Block 256 thr = 8 waves, tile 64(M) x 128(N), K step 32. f32 accumulate.
// Output: f16 to C16, or f32 to C32 (logits) when C32 != null.
__global__ __launch_bounds__(256) void wmma_gemm(
    const _Float16* __restrict__ A, int lda,
    const _Float16* __restrict__ Bt, int ldb,
    const float* __restrict__ bias,
    _Float16* __restrict__ C16, float* __restrict__ C32, int ldc,
    int Kd, float alpha, int act)
{
    __shared__ __align__(16) _Float16 As[64][40];     // [m][k], 80B pitch
    __shared__ __align__(16) _Float16 Bs_t[128][40];  // [n][k], 80B pitch

    int tid  = threadIdx.x;
    int lane = tid & 31;
    int wid  = tid >> 5;
    int wm   = wid & 3, wn = wid >> 2;
    int m0   = blockIdx.y * 64;
    int n0   = blockIdx.x * 128;
    int mloc = lane & 15;
    int hi   = (lane >> 4) & 1;

    v8f z = {};
    v8f acc[4];
#pragma unroll
    for (int f = 0; f < 4; ++f) acc[f] = z;

    for (int k0 = 0; k0 < Kd; k0 += 32) {
        {   // A tile 64x32: one h8 (16B) load+store per thread
            int r = tid >> 2, c = (tid & 3) << 3;
            *(h8*)&As[r][c] = *(const h8*)&A[(size_t)(m0 + r) * lda + k0 + c];
        }
#pragma unroll
        for (int i = 0; i < 2; ++i) {                 // B tile 128x32: 2 h8 per thread
            int idx = tid + i * 256;
            int nn = idx >> 2, kc = (idx & 3) << 3;
            *(h8*)&Bs_t[nn][kc] = *(const h8*)&Bt[(size_t)(n0 + nn) * ldb + k0 + kc];
        }
        if (k0 + 32 < Kd)
            __builtin_prefetch(&A[(size_t)(m0 + (tid & 63)) * lda + k0 + 32], 0, 3);
        __syncthreads();

        const _Float16* arow = &As[wm * 16 + mloc][0];
        h8 a0 = *(const h8*)&arow[hi * 8];
        h8 a1 = *(const h8*)&arow[16 + hi * 8];
        v16h a = frag_cat(a0, a1);
#pragma unroll
        for (int f = 0; f < 4; ++f) {
            const _Float16* brow = &Bs_t[wn * 64 + f * 16 + mloc][0];
            h8 b0 = *(const h8*)&brow[hi * 8];
            h8 b1 = *(const h8*)&brow[16 + hi * 8];
            v16h b = frag_cat(b0, b1);
            acc[f] = __builtin_amdgcn_wmma_f32_16x16x32_f16(
                         false, a, false, b, (short)0, acc[f], false, false);
        }
        __syncthreads();
    }

#pragma unroll
    for (int f = 0; f < 4; ++f) {
        int col = n0 + wn * 64 + f * 16 + mloc;
        float bb = bias ? bias[col] : 0.0f;
#pragma unroll
        for (int r = 0; r < 8; ++r) {
            int row = m0 + wm * 16 + r + 8 * hi;
            float v = acc[f][r] * alpha + bb;
            if (act == ACT_GELU) v = 0.5f * v * (1.0f + erff(v * 0.70710678118f));
            if (C32) C32[(size_t)row * ldc + col] = v;
            else     C16[(size_t)row * ldc + col] = (_Float16)v;
        }
    }
}

// ---------------------------------------------------------------------------
// Fused flash attention (dh=64, heads=8), f16 in / f16 out, f32 accumulate.
// Block = 4 waves x 16 query rows; online softmax; 1/sqrt(dh) folded into scores.
__global__ __launch_bounds__(128) void flash_attn(
    const _Float16* __restrict__ qb, int ldq,
    const _Float16* __restrict__ kb, int ldk,
    const _Float16* __restrict__ vb, int ldv,
    _Float16* __restrict__ ob, int ldo)
{
    __shared__ __align__(16) _Float16 Kt[32][72];      // [key][dh], 144B pitch
    __shared__ __align__(16) _Float16 Vt_t[64][40];    // [dh][key], 80B pitch
    __shared__ __align__(16) _Float16 Pt[4][16][40];   // [wave][m][key]

    int tid  = threadIdx.x;
    int lane = tid & 31;
    int wid  = tid >> 5;
    int n    = blockIdx.y >> 3;
    int h    = blockIdx.y & 7;
    int q0   = blockIdx.x * 64 + wid * 16;
    int hc   = h * DH;
    int mloc = lane & 15;
    int hi   = (lane >> 4) & 1;

    // Q 16x64 as two A-frags: two b128 global loads each
    const _Float16* qrow = qb + (size_t)(n * S_DIM + q0 + mloc) * ldq + hc;
    v16h aq[2];
#pragma unroll
    for (int f = 0; f < 2; ++f) {
        h8 lo  = *(const h8*)&qrow[f * 32 + hi * 8];
        h8 hi8 = *(const h8*)&qrow[f * 32 + 16 + hi * 8];
        aq[f] = frag_cat(lo, hi8);
    }

    float mi[8], li[8];
#pragma unroll
    for (int r = 0; r < 8; ++r) { mi[r] = -3.0e38f; li[r] = 0.0f; }
    v8f z = {};
    v8f oacc[4];
#pragma unroll
    for (int d = 0; d < 4; ++d) oacc[d] = z;

    for (int kt = 0; kt < S_DIM; kt += 32) {
#pragma unroll
        for (int i = 0; i < 2; ++i) {                 // K/V tiles 32x64 f16
            int idx = tid + i * 128;                  // 256 h8 chunks each
            int r = idx >> 3, c = (idx & 7) << 3;
            *(h8*)&Kt[r][c] = *(const h8*)&kb[(size_t)(n * S_DIM + kt + r) * ldk + hc + c];
            h8 hv = *(const h8*)&vb[(size_t)(n * S_DIM + kt + r) * ldv + hc + c];
#pragma unroll
            for (int j = 0; j < 8; ++j) Vt_t[c + j][r] = hv[j];
        }
        __syncthreads();

        v8f sc[2];                                    // raw scores (x sqrt(dh))
#pragma unroll
        for (int g = 0; g < 2; ++g) {
            v8f cacc = z;
            const _Float16* krow = &Kt[g * 16 + mloc][0];
#pragma unroll
            for (int f = 0; f < 2; ++f) {
                h8 b0 = *(const h8*)&krow[f * 32 + hi * 8];
                h8 b1 = *(const h8*)&krow[f * 32 + 16 + hi * 8];
                v16h b = frag_cat(b0, b1);
                cacc = __builtin_amdgcn_wmma_f32_16x16x32_f16(
                           false, aq[f], false, b, (short)0, cacc, false, false);
            }
            sc[g] = cacc;
        }

        float p0[8], p1[8];                           // online softmax (scale 1/8)
#pragma unroll
        for (int r = 0; r < 8; ++r) {
            float s0 = sc[0][r] * 0.125f;
            float s1 = sc[1][r] * 0.125f;
            float v = fmaxf(s0, s1);
#pragma unroll
            for (int msk = 1; msk < 16; msk <<= 1) v = fmaxf(v, __shfl_xor(v, msk, 32));
            float mnew = fmaxf(mi[r], v);
            float corr = __expf(mi[r] - mnew);
            mi[r] = mnew;
            p0[r] = __expf(s0 - mnew);
            p1[r] = __expf(s1 - mnew);
            float rs = p0[r] + p1[r];
#pragma unroll
            for (int msk = 1; msk < 16; msk <<= 1) rs += __shfl_xor(rs, msk, 32);
            li[r] = li[r] * corr + rs;
#pragma unroll
            for (int d = 0; d < 4; ++d) oacc[d][r] *= corr;
        }

#pragma unroll
        for (int r = 0; r < 8; ++r) {                 // stage P (C-layout -> LDS)
            Pt[wid][r + 8 * hi][mloc]      = (_Float16)p0[r];
            Pt[wid][r + 8 * hi][16 + mloc] = (_Float16)p1[r];
        }
        asm volatile("s_wait_dscnt 0" ::: "memory");  // in-wave LDS RAW fence
        const _Float16* prow = &Pt[wid][mloc][0];     // reload P as A-frag 16x32
        h8 ap0 = *(const h8*)&prow[hi * 8];
        h8 ap1 = *(const h8*)&prow[16 + hi * 8];
        v16h ap = frag_cat(ap0, ap1);

#pragma unroll
        for (int d = 0; d < 4; ++d) {                 // O += P @ V
            const _Float16* vrow = &Vt_t[d * 16 + mloc][0];
            h8 bv0 = *(const h8*)&vrow[hi * 8];
            h8 bv1 = *(const h8*)&vrow[16 + hi * 8];
            v16h bv = frag_cat(bv0, bv1);
            oacc[d] = __builtin_amdgcn_wmma_f32_16x16x32_f16(
                          false, ap, false, bv, (short)0, oacc[d], false, false);
        }
        __syncthreads();
    }

#pragma unroll
    for (int d = 0; d < 4; ++d) {
        int col = hc + d * 16 + mloc;
#pragma unroll
        for (int r = 0; r < 8; ++r) {
            int row = q0 + r + 8 * hi;
            ob[(size_t)(n * S_DIM + row) * ldo + col] = (_Float16)(oacc[d][r] / li[r]);
        }
    }
}

// ---------------------------------------------------------------------------
// out = LayerNorm(a + b) * gamma + beta   (f16 in/out, f32 statistics)
__global__ __launch_bounds__(256) void ln_residual(
    const _Float16* __restrict__ a, const _Float16* __restrict__ b,
    const float* __restrict__ gam, const float* __restrict__ bet,
    _Float16* __restrict__ out)
{
    __shared__ float red[256];
    __shared__ float stat[2];
    int row = blockIdx.x, t = threadIdx.x;
    size_t base = (size_t)row * C_DIM;
    float x0 = (float)a[base + t]       + (float)b[base + t];
    float x1 = (float)a[base + 256 + t] + (float)b[base + 256 + t];
    red[t] = x0 + x1;
    __syncthreads();
    for (int off = 128; off > 0; off >>= 1) {
        if (t < off) red[t] += red[t + off];
        __syncthreads();
    }
    if (t == 0) stat[0] = red[0] * (1.0f / 512.0f);
    __syncthreads();
    float mu = stat[0];
    float d0 = x0 - mu, d1 = x1 - mu;
    red[t] = d0 * d0 + d1 * d1;
    __syncthreads();
    for (int off = 128; off > 0; off >>= 1) {
        if (t < off) red[t] += red[t + off];
        __syncthreads();
    }
    if (t == 0) stat[1] = rsqrtf(red[0] * (1.0f / 512.0f) + 1e-5f);
    __syncthreads();
    float rs = stat[1];
    out[base + t]       = (_Float16)(d0 * rs * gam[t]       + bet[t]);
    out[base + 256 + t] = (_Float16)(d1 * rs * gam[256 + t] + bet[256 + t]);
}

// ---------------------------------------------------------------------------
// idx = argmax(logit + gumbel); code; quantized = value[idx] + PE (f16)
__global__ __launch_bounds__(256) void gumbel_argmax_gather(
    const float* __restrict__ logit, const float* __restrict__ ug,
    const _Float16* __restrict__ valb, int* __restrict__ code,
    _Float16* __restrict__ quant)
{
    __shared__ float smax[256];
    __shared__ int   sidx[256];
    int row = blockIdx.x, t = threadIdx.x;
    size_t base = (size_t)row * K_CODE;
    float best = -3.0e38f; int bidx = 0;
    for (int k = t; k < K_CODE; k += 256) {
        float u = ug[base + k];
        float g = -__logf(-__logf(u + 1e-10f) + 1e-10f);
        float zv = logit[base + k] + g;
        if (zv > best) { best = zv; bidx = k; }
    }
    smax[t] = best; sidx[t] = bidx;
    __syncthreads();
    for (int off = 128; off > 0; off >>= 1) {
        if (t < off) {
            float o = smax[t + off]; int oi = sidx[t + off];
            if (o > smax[t] || (o == smax[t] && oi < sidx[t])) { smax[t] = o; sidx[t] = oi; }
        }
        __syncthreads();
    }
    int idx = sidx[0];
    if (t == 0) code[row] = idx;
    int s = row & (S_DIM - 1);
    quant[(size_t)row * C_DIM + t] =
        (_Float16)((float)valb[(size_t)idx * C_DIM + t] + pe_val(s, t));
    quant[(size_t)row * C_DIM + 256 + t] =
        (_Float16)((float)valb[(size_t)idx * C_DIM + 256 + t] + pe_val(s, 256 + t));
}

// ---------------------------------------------------------------------------
// (N,S,C) f16 -> (N,C,H,W) f32
__global__ void out_transpose(const _Float16* __restrict__ y, float* __restrict__ out) {
    int idx = blockIdx.x * blockDim.x + threadIdx.x;  // over N*C*S
    int s   = idx & (S_DIM - 1);
    int rem = idx >> 10;
    int c   = rem & (C_DIM - 1);
    int n   = rem >> 9;
    out[idx] = (float)y[(size_t)(n * S_DIM + s) * C_DIM + c];
}

// ---------------------------------------------------------------------------
extern "C" void kernel_launch(void* const* d_in, const int* in_sizes, int n_in,
                              void* d_out, int out_size, void* d_ws, size_t ws_size,
                              hipStream_t stream)
{
    const float* latent0      = (const float*)d_in[0];
    const float* ug           = (const float*)d_in[1];
    const float* enc_qkv_w    = (const float*)d_in[2];
    const float* enc_qkv_b    = (const float*)d_in[3];
    const float* enc_out_w    = (const float*)d_in[4];
    const float* enc_out_b    = (const float*)d_in[5];
    const float* enc_ff1_w    = (const float*)d_in[6];
    const float* enc_ff1_b    = (const float*)d_in[7];
    const float* enc_ff2_w    = (const float*)d_in[8];
    const float* enc_ff2_b    = (const float*)d_in[9];
    const float* enc_ln1_s    = (const float*)d_in[10];
    const float* enc_ln1_b    = (const float*)d_in[11];
    const float* enc_ln2_s    = (const float*)d_in[12];
    const float* enc_ln2_b    = (const float*)d_in[13];
    const float* dec_sa_qkv_w = (const float*)d_in[14];
    const float* dec_sa_qkv_b = (const float*)d_in[15];
    const float* dec_sa_out_w = (const float*)d_in[16];
    const float* dec_sa_out_b = (const float*)d_in[17];
    const float* dec_ca_qkv_w = (const float*)d_in[18];
    const float* dec_ca_qkv_b = (const float*)d_in[19];
    const float* dec_ca_out_w = (const float*)d_in[20];
    const float* dec_ca_out_b = (const float*)d_in[21];
    const float* dec_ff1_w    = (const float*)d_in[22];
    const float* dec_ff1_b    = (const float*)d_in[23];
    const float* dec_ff2_w    = (const float*)d_in[24];
    const float* dec_ff2_b    = (const float*)d_in[25];
    const float* dec_ln1_s    = (const float*)d_in[26];
    const float* dec_ln1_b    = (const float*)d_in[27];
    const float* dec_ln2_s    = (const float*)d_in[28];
    const float* dec_ln2_b    = (const float*)d_in[29];
    const float* dec_ln3_s    = (const float*)d_in[30];
    const float* dec_ln3_b    = (const float*)d_in[31];
    const float* codebook0    = (const float*)d_in[32];
    const float* q_w          = (const float*)d_in[33];
    const float* q_b          = (const float*)d_in[34];
    const float* k_w          = (const float*)d_in[35];
    const float* k_b          = (const float*)d_in[36];
    const float* v_w          = (const float*)d_in[37];
    const float* v_b          = (const float*)d_in[38];

    // workspace layout (half elements)
    _Float16* W = (_Float16*)d_ws;
    _Float16* Wencqkv  = W;                         //  786432 (1536x512)
    _Float16* Wencout  = W + (size_t)786432;        //  262144 (512x512)
    _Float16* Wencff1  = W + (size_t)1048576;
    _Float16* Wencff2  = W + (size_t)1310720;
    _Float16* Wsaqkv   = W + (size_t)1572864;       //  786432
    _Float16* Wsaout   = W + (size_t)2359296;
    _Float16* Wcaqkv   = W + (size_t)2621440;       //  786432
    _Float16* Wcaout   = W + (size_t)3407872;
    _Float16* Wdff1    = W + (size_t)3670016;
    _Float16* Wdff2    = W + (size_t)3932160;
    _Float16* Wq       = W + (size_t)4194304;
    _Float16* Wk       = W + (size_t)4456448;
    _Float16* Wv       = W + (size_t)4718592;
    _Float16* CBh      = W + (size_t)4980736;       // 2097152 (4096x512, row-major)
    _Float16* KEYB     = W + (size_t)7077888;       // 2097152 (4096x512 = NxK)
    _Float16* VALB     = W + (size_t)9175040;       // 2097152
    _Float16* X        = W + (size_t)11272192;      // 16384x512
    _Float16* QKV      = W + (size_t)19660800;      // 16384x1536
    _Float16* T0       = W + (size_t)44826624;      // 16384x512
    _Float16* T1       = W + (size_t)53215232;
    _Float16* T2       = W + (size_t)61603840;
    _Float16* QT       = W + (size_t)69992448;      // quantized tokens

    // d_out layout: out(8388608 f32) | code(16384 int) | logit(67108864 f32)
    float* out_img = (float*)d_out;
    int*   code    = (int*)((char*)d_out + (size_t)8388608 * 4);
    float* logit   = (float*)((char*)d_out + (size_t)8404992 * 4);

    dim3 gN512(4, 256), gN1024(8, 256), gN1536(12, 256), gAttn(16, 128);

    // ---- one-time weight prep: f16, transposed to (N x K) ----
    cvt_transpose_w<<<3072, 256, 0, stream>>>(enc_qkv_w,    Wencqkv, 512, 1536);
    cvt_transpose_w<<<1024, 256, 0, stream>>>(enc_out_w,    Wencout, 512, 512);
    cvt_transpose_w<<<1024, 256, 0, stream>>>(enc_ff1_w,    Wencff1, 512, 512);
    cvt_transpose_w<<<1024, 256, 0, stream>>>(enc_ff2_w,    Wencff2, 512, 512);
    cvt_transpose_w<<<3072, 256, 0, stream>>>(dec_sa_qkv_w, Wsaqkv,  512, 1536);
    cvt_transpose_w<<<1024, 256, 0, stream>>>(dec_sa_out_w, Wsaout,  512, 512);
    cvt_transpose_w<<<3072, 256, 0, stream>>>(dec_ca_qkv_w, Wcaqkv,  512, 1536);
    cvt_transpose_w<<<1024, 256, 0, stream>>>(dec_ca_out_w, Wcaout,  512, 512);
    cvt_transpose_w<<<1024, 256, 0, stream>>>(dec_ff1_w,    Wdff1,   512, 512);
    cvt_transpose_w<<<1024, 256, 0, stream>>>(dec_ff2_w,    Wdff2,   512, 512);
    cvt_transpose_w<<<1024, 256, 0, stream>>>(q_w,          Wq,      512, 512);
    cvt_transpose_w<<<1024, 256, 0, stream>>>(k_w,          Wk,      512, 512);
    cvt_transpose_w<<<1024, 256, 0, stream>>>(v_w,          Wv,      512, 512);
    cvt_plain<<<8192, 256, 0, stream>>>(codebook0, CBh);

    // ---- encoder ----
    embed_kernel<<<32768, 256, 0, stream>>>(latent0, X);
    wmma_gemm<<<gN1536, 256, 0, stream>>>(X, 512, Wencqkv, 512, enc_qkv_b,
                                          QKV, nullptr, 1536, 512, 1.0f, ACT_NONE);
    flash_attn<<<gAttn, 128, 0, stream>>>(QKV, 1536, QKV + 512, 1536, QKV + 1024, 1536, T0, 512);
    wmma_gemm<<<gN512, 256, 0, stream>>>(T0, 512, Wencout, 512, enc_out_b,
                                         T2, nullptr, 512, 512, 1.0f, ACT_NONE);
    ln_residual<<<NS, 256, 0, stream>>>(X, T2, enc_ln1_s, enc_ln1_b, T1);
    wmma_gemm<<<gN512, 256, 0, stream>>>(T1, 512, Wencff1, 512, enc_ff1_b,
                                         T0, nullptr, 512, 512, 1.0f, ACT_GELU);
    wmma_gemm<<<gN512, 256, 0, stream>>>(T0, 512, Wencff2, 512, enc_ff2_b,
                                         T2, nullptr, 512, 512, 1.0f, ACT_NONE);
    ln_residual<<<NS, 256, 0, stream>>>(T1, T2, enc_ln2_s, enc_ln2_b, X);   // X = x2

    // ---- quantizer ----
    wmma_gemm<<<gN512, 256, 0, stream>>>(X, 512, Wq, 512, q_b,
                                         T0, nullptr, 512, 512, 1.0f, ACT_NONE);  // query
    wmma_gemm<<<dim3(4, 64), 256, 0, stream>>>(CBh, 512, Wk, 512, k_b,
                                               KEYB, nullptr, 512, 512, 1.0f, ACT_NONE);
    wmma_gemm<<<dim3(4, 64), 256, 0, stream>>>(CBh, 512, Wv, 512, v_b,
                                               VALB, nullptr, 512, 512, 1.0f, ACT_NONE);
    wmma_gemm<<<dim3(32, 256), 256, 0, stream>>>(T0, 512, KEYB, 512, nullptr,
                                                 nullptr, logit, 4096, 512,
                                                 1.0f / 64.0f, ACT_NONE);         // /sqrt(K)
    gumbel_argmax_gather<<<NS, 256, 0, stream>>>(logit, ug, VALB, code, QT);

    // ---- decoder: self-attention ----
    wmma_gemm<<<gN1536, 256, 0, stream>>>(QT, 512, Wsaqkv, 512, dec_sa_qkv_b,
                                          QKV, nullptr, 1536, 512, 1.0f, ACT_NONE);
    flash_attn<<<gAttn, 128, 0, stream>>>(QKV, 1536, QKV + 512, 1536, QKV + 1024, 1536, T0, 512);
    wmma_gemm<<<gN512, 256, 0, stream>>>(T0, 512, Wsaout, 512, dec_sa_out_b,
                                         T2, nullptr, 512, 512, 1.0f, ACT_NONE);
    ln_residual<<<NS, 256, 0, stream>>>(QT, T2, dec_ln1_s, dec_ln1_b, T1);  // T1 = y1

    // ---- decoder: cross-attention (mem == quantized) ----
    wmma_gemm<<<gN512, 256, 0, stream>>>(T1, 512, Wcaqkv, 512, dec_ca_qkv_b,
                                         T2, nullptr, 512, 512, 1.0f, ACT_NONE);          // q
    wmma_gemm<<<gN1024, 256, 0, stream>>>(QT, 512, Wcaqkv + (size_t)512 * 512, 512,
                                          dec_ca_qkv_b + 512, QKV, nullptr, 1024,
                                          512, 1.0f, ACT_NONE);                           // k|v
    flash_attn<<<gAttn, 128, 0, stream>>>(T2, 512, QKV, 1024, QKV + 512, 1024, T0, 512);
    wmma_gemm<<<gN512, 256, 0, stream>>>(T0, 512, Wcaout, 512, dec_ca_out_b,
                                         T2, nullptr, 512, 512, 1.0f, ACT_NONE);
    ln_residual<<<NS, 256, 0, stream>>>(T1, T2, dec_ln2_s, dec_ln2_b, X);   // X = y2

    // ---- decoder: FFN ----
    wmma_gemm<<<gN512, 256, 0, stream>>>(X, 512, Wdff1, 512, dec_ff1_b,
                                         T0, nullptr, 512, 512, 1.0f, ACT_GELU);
    wmma_gemm<<<gN512, 256, 0, stream>>>(T0, 512, Wdff2, 512, dec_ff2_b,
                                         T2, nullptr, 512, 512, 1.0f, ACT_NONE);
    ln_residual<<<NS, 256, 0, stream>>>(X, T2, dec_ln3_s, dec_ln3_b, T1);   // T1 = y3

    out_transpose<<<32768, 256, 0, stream>>>(T1, out_img);
}
